// MultiHeadSelfAttention_65025804862080
// MI455X (gfx1250) — matrix operand order repaired and checked by reading, not compiled, not run on hardware
//
#include <hip/hip_runtime.h>
#include <hip/hip_bf16.h>

typedef __attribute__((ext_vector_type(16))) _Float16 v16h;
typedef __attribute__((ext_vector_type(8)))  _Float16 v8h;
typedef __attribute__((ext_vector_type(8)))  float    v8f;

#define WMMA_F32_F16(a, b, c) \
  __builtin_amdgcn_wmma_f32_16x16x32_f16(false, (a), false, (b), (short)0, (c), false, false)

union ABFrag { v16h v; v8h h[2]; };

// A-fragment (16xK tile, K=32 slice) from row-major [M, ld] f16.
// Lane L: row m0+L%16; halves K = [k0+(L/16)*8, +8) and [k0+16+(L/16)*8, +8).
__device__ inline v16h load_a_frag(const _Float16* __restrict__ base, int ld,
                                   int m0, int k0, int lane) {
  const _Float16* p = base + (size_t)(m0 + (lane & 15)) * ld + k0 + ((lane >> 4) << 3);
  ABFrag f;
  f.h[0] = *(const v8h*)p;
  f.h[1] = *(const v8h*)(p + 16);
  return f.v;
}

// B-fragment (Kx16 tile, K=32 slice) from row-major [N, ld] f16 (i.e. B^T).
// Lane L: row n0+L%16; halves K = [k0+(L/16)*8, +8) and [k0+16+(L/16)*8, +8).
__device__ inline v16h load_b_frag(const _Float16* __restrict__ base, int ld,
                                   int n0, int k0, int lane) {
  const _Float16* p = base + (size_t)(n0 + (lane & 15)) * ld + k0 + ((lane >> 4) << 3);
  ABFrag f;
  f.h[0] = *(const v8h*)p;
  f.h[1] = *(const v8h*)(p + 16);
  return f.v;
}

__global__ void cvt_f32_f16(const float* __restrict__ in, _Float16* __restrict__ out, int n) {
  int i = blockIdx.x * blockDim.x + threadIdx.x;
  int stride = gridDim.x * blockDim.x;
  for (; i < n; i += stride) out[i] = (_Float16)in[i];
}

enum { MODE_ROPE = 0, MODE_V = 1, MODE_OUT = 2 };

// C = A(row-major [M,K]) * B^T (B row-major [N,K]) + bias, with fused epilogues.
// One wave -> 16x64 output tile. Block = 4 waves (4 m-tiles), grid = (M/64, N/64).
template <int MODE>
__global__ __launch_bounds__(128) void gemm_nt(
    const _Float16* __restrict__ A, const _Float16* __restrict__ B,
    const float* __restrict__ bias,
    const float* __restrict__ cosT, const float* __restrict__ sinT,
    _Float16* __restrict__ outH, float* __restrict__ outF,
    int M, int N, int K) {
  const int lane = threadIdx.x & 31;
  const int wave = threadIdx.x >> 5;
  const int m0 = (blockIdx.x * 4 + wave) * 16;
  const int n0 = blockIdx.y * 64;
  if (m0 >= M) return;

  v8f acc[4] = {};
  for (int k0 = 0; k0 < K; k0 += 32) {
    v16h a = load_a_frag(A, K, m0, k0, lane);
#pragma unroll
    for (int t = 0; t < 4; ++t) {
      v16h b = load_b_frag(B, K, n0 + t * 16, k0, lane);
      acc[t] = WMMA_F32_F16(a, b, acc[t]);
    }
  }

  // D layout: lane L holds col n0+t*16+L%16; VGPR r holds row m0 + r + 8*(L/16).
  const int mrow = (lane >> 4) << 3;
#pragma unroll
  for (int t = 0; t < 4; ++t) {
    const int col = n0 + t * 16 + (lane & 15);
#pragma unroll
    for (int r = 0; r < 8; ++r) {
      const int row = m0 + mrow + r;
      float v = acc[t][r] + bias[col];
      if constexpr (MODE == MODE_ROPE) {
        // RoPE: out[j] = v[j]*cos[j%64] + (j even ? -v[j^1] : v[j^1]) * sin[j%64]
        // partner col = col^1 lives in lane^1 (same row, same half).
        float pv = __shfl_xor(v, 1, 32);
        int dp = col & 63;
        float c = cosT[dp], s = sinT[dp];
        float res = v * c + ((col & 1) ? pv * s : -pv * s);
        outH[(size_t)row * N + col] = (_Float16)res;
      } else if constexpr (MODE == MODE_V) {
        // Scatter into per-head transposed layout Vt[b*16+h][d=64][t'=2048].
        int bb = row >> 11, rb = row & 2047;
        int hh = rb >> 7, sub = rb & 127;
        int chunk = col >> 6, dd = col & 63;
        int tprime = sub * 16 + chunk;
        size_t idx = ((size_t)(bb * 16 + hh) * 64 + dd) * 2048 + tprime;
        outH[idx] = (_Float16)v;
      } else {
        outF[(size_t)row * N + col] = v;
      }
    }
  }
}

// Flash-style attention. One wave = 16 q rows, online softmax over 32-key tiles.
// Qh/Kh: per-head contiguous slabs [bh][2048][64] f16 (= flat [4096,1024]).
// Vt: [bh][64][2048] f16. Output: Attn f16 [4096,1024] at (b*2048+t', h*64+d').
__global__ __launch_bounds__(128) void attn_fwd(
    const _Float16* __restrict__ Qh, const _Float16* __restrict__ Kh,
    const _Float16* __restrict__ Vt, _Float16* __restrict__ attnOut) {
  __shared__ __align__(32) _Float16 lds[4][16 * 32];
  const int lane = threadIdx.x & 31;
  const int wave = threadIdx.x >> 5;
  const int bh = blockIdx.y;
  const int b = bh >> 4, h = bh & 15;
  const _Float16* Q = Qh + (size_t)bh * 2048 * 64;
  const _Float16* K = Kh + (size_t)bh * 2048 * 64;
  const _Float16* V = Vt + (size_t)bh * 64 * 2048;
  const int m0 = blockIdx.x * 64 + wave * 16;

  // Q A-fragments for d=0..63 stay in registers for the whole pass.
  v16h qa0 = load_a_frag(Q, 64, m0, 0, lane);
  v16h qa1 = load_a_frag(Q, 64, m0, 32, lane);

  v8f o[4] = {};
  float rm[8], rl[8];
#pragma unroll
  for (int r = 0; r < 8; ++r) { rm[r] = -1e30f; rl[r] = 0.f; }
  const float scale = 0.125f;  // 1/sqrt(64)

  for (int kt = 0; kt < 2048; kt += 32) {
    v8f s0 = {}, s1 = {};
    {
      v16h b0 = load_b_frag(K, 64, kt, 0, lane);
      v16h b1 = load_b_frag(K, 64, kt, 32, lane);
      s0 = WMMA_F32_F16(qa0, b0, s0);
      s0 = WMMA_F32_F16(qa1, b1, s0);
      v16h b2 = load_b_frag(K, 64, kt + 16, 0, lane);
      v16h b3 = load_b_frag(K, 64, kt + 16, 32, lane);
      s1 = WMMA_F32_F16(qa0, b2, s1);
      s1 = WMMA_F32_F16(qa1, b3, s1);
    }
    _Float16* P = lds[wave];
#pragma unroll
    for (int r = 0; r < 8; ++r) {
      float a0 = s0[r] * scale;
      float a1 = s1[r] * scale;
      float tmax = fmaxf(a0, a1);
#pragma unroll
      for (int msk = 1; msk <= 8; msk <<= 1)
        tmax = fmaxf(tmax, __shfl_xor(tmax, msk, 32));
      float nm = fmaxf(rm[r], tmax);
      float corr = __expf(rm[r] - nm);
      float p0 = __expf(a0 - nm);
      float p1 = __expf(a1 - nm);
      float rsum = p0 + p1;
#pragma unroll
      for (int msk = 1; msk <= 8; msk <<= 1)
        rsum += __shfl_xor(rsum, msk, 32);
      rl[r] = rl[r] * corr + rsum;
      rm[r] = nm;
#pragma unroll
      for (int t = 0; t < 4; ++t) o[t][r] *= corr;
      // Stage P tile [16 q][32 k] row-major in LDS for A-fragment reload.
      int m = r + ((lane >> 4) << 3);
      P[m * 32 + (lane & 15)] = (_Float16)p0;
      P[m * 32 + 16 + (lane & 15)] = (_Float16)p1;
    }
    asm volatile("s_wait_dscnt 0x0" ::: "memory");  // same-wave LDS RAW
    ABFrag pa;
    const _Float16* lp = &P[(lane & 15) * 32 + ((lane >> 4) << 3)];
    pa.h[0] = *(const v8h*)lp;
    pa.h[1] = *(const v8h*)(lp + 16);
#pragma unroll
    for (int t = 0; t < 4; ++t) {
      v16h vb = load_b_frag(V, 2048, t * 16, kt, lane);
      o[t] = WMMA_F32_F16(pa.v, vb, o[t]);
    }
  }

  const int mrow = (lane >> 4) << 3;
#pragma unroll
  for (int r = 0; r < 8; ++r) {
    float inv = 1.f / rl[r];
    int tprime = m0 + mrow + r;
    size_t rowbase = (size_t)(b * 2048 + tprime) * 1024;
#pragma unroll
    for (int t = 0; t < 4; ++t) {
      int col = h * 64 + t * 16 + (lane & 15);
      attnOut[rowbase + col] = (_Float16)(o[t][r] * inv);
    }
  }
}

extern "C" void kernel_launch(void* const* d_in, const int* in_sizes, int n_in,
                              void* d_out, int out_size, void* d_ws, size_t ws_size,
                              hipStream_t stream) {
  const float* x  = (const float*)d_in[0];
  const float* wq = (const float*)d_in[1];
  const float* bq = (const float*)d_in[2];
  const float* wk = (const float*)d_in[3];
  const float* bk = (const float*)d_in[4];
  const float* wv = (const float*)d_in[5];
  const float* bv = (const float*)d_in[6];
  const float* wo = (const float*)d_in[7];
  const float* bo = (const float*)d_in[8];
  const float* cosT = (const float*)d_in[9]  + 2048 * 64;  // reference indexes cos[T]
  const float* sinT = (const float*)d_in[10] + 2048 * 64;
  float* out = (float*)d_out;

  _Float16* p = (_Float16*)d_ws;
  _Float16* Xh  = p; p += (size_t)4096 * 1024;
  _Float16* WqH = p; p += (size_t)1024 * 1024;
  _Float16* WkH = p; p += (size_t)1024 * 1024;
  _Float16* WvH = p; p += (size_t)1024 * 1024;
  _Float16* WoH = p; p += (size_t)1024 * 1024;
  _Float16* QhH = p; p += (size_t)4096 * 1024;
  _Float16* KhH = p; p += (size_t)4096 * 1024;
  _Float16* VtH = p; p += (size_t)4096 * 1024;
  _Float16* AtH = p; p += (size_t)4096 * 1024;   // 48 MB total

  cvt_f32_f16<<<512, 256, 0, stream>>>(x,  Xh,  4096 * 1024);
  cvt_f32_f16<<<256, 256, 0, stream>>>(wq, WqH, 1024 * 1024);
  cvt_f32_f16<<<256, 256, 0, stream>>>(wk, WkH, 1024 * 1024);
  cvt_f32_f16<<<256, 256, 0, stream>>>(wv, WvH, 1024 * 1024);
  cvt_f32_f16<<<256, 256, 0, stream>>>(wo, WoH, 1024 * 1024);

  dim3 gg(64, 16), bb(128);
  gemm_nt<MODE_ROPE><<<gg, bb, 0, stream>>>(Xh, WqH, bq, cosT, sinT, QhH, nullptr, 4096, 1024, 1024);
  gemm_nt<MODE_ROPE><<<gg, bb, 0, stream>>>(Xh, WkH, bk, cosT, sinT, KhH, nullptr, 4096, 1024, 1024);
  gemm_nt<MODE_V>   <<<gg, bb, 0, stream>>>(Xh, WvH, bv, nullptr, nullptr, VtH, nullptr, 4096, 1024, 1024);

  attn_fwd<<<dim3(32, 32), bb, 0, stream>>>(QhH, KhH, VtH, AtH);

  gemm_nt<MODE_OUT><<<gg, bb, 0, stream>>>(AtH, WoH, bo, nullptr, nullptr, nullptr, out, 4096, 1024, 1024);
}
